// MultiHeadAttention_34213709480300
// MI455X (gfx1250) — matrix operand order repaired and checked by reading, compiled, not run on hardware
//
#include <hip/hip_runtime.h>

// ---------------------------------------------------------------------------
// MHA forward for MI455X (gfx1250, wave32, WMMA 16x16x32 bf16).
// Pipeline: f32->bf16 convert | Q/K/V projections (WMMA GEMM 64x64/wave,
// scattered epilogues) | flash-attention computed TRANSPOSED
// (S^T = K*Q^T, O^T = V^T*P^T) so the softmax axis is in-lane: per-lane
// scalar stats, 1 shuffle per reduction, vector mask loads, packed LDS
// transpose of P^T | output projection.
// ---------------------------------------------------------------------------

typedef __attribute__((ext_vector_type(16))) __bf16 v16bf;
typedef __attribute__((ext_vector_type(8)))  float  v8f;

#define HID  1024
#define NH   16
#define HD   64
#define BB   2
#define SS   2048
#define MTOT (BB * SS)   // 4096 rows

union FragBF { uint4 u[2]; v16bf v; };
union PackBF { unsigned short us[8]; uint4 u; };

__device__ __forceinline__ unsigned short f2bf(float f) {
  unsigned int u = __float_as_uint(f);
  unsigned int r = u + 0x7FFFu + ((u >> 16) & 1u);   // round-to-nearest-even
  return (unsigned short)(r >> 16);
}

// -------------------------- f32 -> bf16 convert ----------------------------
__global__ void __launch_bounds__(256)
cvt_f32_bf16(const float* __restrict__ in, unsigned short* __restrict__ out, int n4) {
  int i = blockIdx.x * blockDim.x + threadIdx.x;
  if (i >= n4) return;
  float4 f = ((const float4*)in)[i];
  uint2 o;
  o.x = (unsigned)f2bf(f.x) | ((unsigned)f2bf(f.y) << 16);
  o.y = (unsigned)f2bf(f.z) | ((unsigned)f2bf(f.w) << 16);
  ((uint2*)out)[i] = o;
}

// ----------------------------- WMMA GEMM -----------------------------------
// C[M,N] = A[M,K](bf16,row-major) * W[N,K](bf16,row-major)^T
// One wave computes a 64(M) x 64(N) tile: 4 A-frags, 4 B-frags, 16 wmma/k-step.
// MODE 0: f32 row-major [M,N]        (final output)
// MODE 1: bf16 [B,NH,S,HD]           (Q, K)
// MODE 2: bf16 [B,NH,HD,S] (V^T), packed b128 stores along S
template <int MODE>
__global__ void __launch_bounds__(256)
gemm_bf16(const unsigned short* __restrict__ A,
          const unsigned short* __restrict__ W,
          void* __restrict__ Cout, int M, int N, int K) {
  const int lane = threadIdx.x & 31;
  const int wid  = (blockIdx.x * blockDim.x + threadIdx.x) >> 5;
  const int ntiles = N >> 6;
  const int m0 = (wid / ntiles) * 64;
  const int n0 = (wid % ntiles) * 64;
  const int r  = lane & 15;
  const int hf = lane >> 4;

  v8f acc[4][4];
#pragma unroll
  for (int mi = 0; mi < 4; ++mi)
#pragma unroll
    for (int ni = 0; ni < 4; ++ni)
      acc[mi][ni] = (v8f){0.f,0.f,0.f,0.f,0.f,0.f,0.f,0.f};

  for (int k0 = 0; k0 < K; k0 += 32) {
    FragBF a[4];
#pragma unroll
    for (int mi = 0; mi < 4; ++mi) {
      // A 16x32 bf16 layout: lane half hf holds K = hf*8..+7 and 16+hf*8..+7
      const unsigned short* ap = A + (size_t)(m0 + mi * 16 + r) * K + k0 + hf * 8;
      a[mi].u[0] = *(const uint4*)(ap);
      a[mi].u[1] = *(const uint4*)(ap + 16);
    }
    FragBF b[4];
#pragma unroll
    for (int ni = 0; ni < 4; ++ni) {
      // B 32x16 bf16 layout: lane half hf holds K = hf*8..+7 and 16+hf*8..+7
      const unsigned short* bp = W + (size_t)(n0 + ni * 16 + r) * K + k0 + hf * 8;
      b[ni].u[0] = *(const uint4*)(bp);
      b[ni].u[1] = *(const uint4*)(bp + 16);
    }
#pragma unroll
    for (int mi = 0; mi < 4; ++mi)
#pragma unroll
      for (int ni = 0; ni < 4; ++ni)
        acc[mi][ni] = __builtin_amdgcn_wmma_f32_16x16x32_bf16(
            false, a[mi].v, false, b[ni].v, (short)0, acc[mi][ni], false, false);
  }

  // C/D layout: lanes 0-15: N=lane, M=vgpr j; lanes 16-31: N=lane-16, M=8+j
#pragma unroll
  for (int mi = 0; mi < 4; ++mi)
#pragma unroll
    for (int ni = 0; ni < 4; ++ni) {
      const int col = n0 + ni * 16 + r;
      const int mb  = m0 + mi * 16 + hf * 8;
      if (MODE == 0) {
        float* C = (float*)Cout;
#pragma unroll
        for (int j = 0; j < 8; ++j)
          C[(size_t)(mb + j) * N + col] = acc[mi][ni][j];
      } else if (MODE == 1) {
        unsigned short* C = (unsigned short*)Cout;
        const int h = col >> 6, d = col & (HD - 1);
#pragma unroll
        for (int j = 0; j < 8; ++j) {
          const int m = mb + j;
          const int bb = m >> 11, s = m & (SS - 1);
          C[(((size_t)bb * NH + h) * SS + s) * HD + d] = f2bf(acc[mi][ni][j]);
        }
      } else {  // MODE 2: V^T [B,NH,HD,S]
        unsigned short* C = (unsigned short*)Cout;
        const int h = col >> 6, d = col & (HD - 1);
        const int bb = mb >> 11, s = mb & (SS - 1);
        PackBF pk;
#pragma unroll
        for (int j = 0; j < 8; ++j) pk.us[j] = f2bf(acc[mi][ni][j]);
        *(uint4*)&C[(((size_t)bb * NH + h) * HD + d) * SS + s] = pk.u;
      }
    }
}

// --------------------------- flash attention (transposed) ------------------
// One wave: 32 queries (two 16-col tiles) of one (b,h); streams keys in
// 32-wide chunks. S^T = K*Q^T and O^T = V^T*P^T keep the softmax axis (keys)
// inside each lane: per-lane scalar running stats, one xor-16 shuffle per
// reduction, vectorized mask loads and packed LDS transpose of P^T.
__global__ void __launch_bounds__(256)
attn_kernel(const unsigned short* __restrict__ Q,    // [B,NH,S,HD] bf16
            const unsigned short* __restrict__ Kmat, // [B,NH,S,HD] bf16
            const unsigned short* __restrict__ Vt,   // [B,NH,HD,S] bf16
            const float* __restrict__ mask,          // [1,1,S,S]  f32
            unsigned short* __restrict__ AO) {       // [MTOT,HID] bf16
  // per-wave P^T staging: [wave][qtile][query(16) x key(32)] bf16
  __shared__ __attribute__((aligned(16))) unsigned short plds[8][2][16 * 32];

  const int lane = threadIdx.x & 31;
  const int wib  = threadIdx.x >> 5;
  const int wg   = blockIdx.x * 8 + wib;
  const int qblocks = SS / 32;                 // 64 query blocks per (b,h)
  const int bh = wg / qblocks;
  const int q0 = (wg % qblocks) * 32;
  const int r  = lane & 15;
  const int hf = lane >> 4;

  const unsigned short* Qp = Q    + (size_t)bh * SS * HD;
  const unsigned short* Kp = Kmat + (size_t)bh * SS * HD;
  const unsigned short* Vp = Vt   + (size_t)bh * HD * SS;

  // Q^T B-fragments: qf[qt][kk]; B 32x16: lane=query col, d chunks kk*32+hf*8 and kk*32+16+hf*8
  FragBF qf[2][2];
#pragma unroll
  for (int qt = 0; qt < 2; ++qt)
#pragma unroll
    for (int kk = 0; kk < 2; ++kk) {
      const unsigned short* qp = Qp + (size_t)(q0 + qt * 16 + r) * HD + kk * 32 + hf * 8;
      qf[qt][kk].u[0] = *(const uint4*)(qp);
      qf[qt][kk].u[1] = *(const uint4*)(qp + 16);
    }

  // O^T accumulators: ot[qt][dt], M=d rows (VGPRs), N=query (lanes)
  v8f ot[2][4];
#pragma unroll
  for (int qt = 0; qt < 2; ++qt)
#pragma unroll
    for (int dt = 0; dt < 4; ++dt) ot[qt][dt] = (v8f){0.f,0.f,0.f,0.f,0.f,0.f,0.f,0.f};
  float mrun[2] = {-3.0e38f, -3.0e38f};
  float lrun[2] = {0.f, 0.f};

  for (int kb = 0; kb < SS; kb += 32) {
    // ---- S^T tiles: st[kt][qt] = K_tile(A) x Q^T(B) ----
    v8f st[2][2];
#pragma unroll
    for (int kt = 0; kt < 2; ++kt)
#pragma unroll
      for (int qt = 0; qt < 2; ++qt) st[kt][qt] = (v8f){0.f,0.f,0.f,0.f,0.f,0.f,0.f,0.f};
#pragma unroll
    for (int kk = 0; kk < 2; ++kk) {
      FragBF ka[2];
#pragma unroll
      for (int kt = 0; kt < 2; ++kt) {
        // A 16x32: lane=key row, d chunks kk*32+hf*8 and kk*32+16+hf*8
        const unsigned short* kp = Kp + (size_t)(kb + kt * 16 + r) * HD + kk * 32 + hf * 8;
        ka[kt].u[0] = *(const uint4*)(kp);
        ka[kt].u[1] = *(const uint4*)(kp + 16);
      }
#pragma unroll
      for (int kt = 0; kt < 2; ++kt)
#pragma unroll
        for (int qt = 0; qt < 2; ++qt)
          st[kt][qt] = __builtin_amdgcn_wmma_f32_16x16x32_bf16(
              false, ka[kt].v, false, qf[qt][kk].v, (short)0, st[kt][qt], false, false);
    }

    // prefetch next chunk of the streamed K / V^T operands
    if (kb + 32 < SS) {
      __builtin_prefetch(Kp + (size_t)(kb + 32 + r) * HD + hf * 16, 0, 1);
      __builtin_prefetch(Vp + (size_t)(lane) * SS + kb + 32, 0, 1);
    }

    // ---- online softmax: keys in-lane, query = lane column ----
#pragma unroll
    for (int qt = 0; qt < 2; ++qt) {
      const int q = q0 + qt * 16 + r;  // this lane's query
      // scores + mask; lane holds keys kb + kt*16 + hf*8 + j (contiguous in j)
      float p[2][8];
      float cm = -3.0e38f;
#pragma unroll
      for (int kt = 0; kt < 2; ++kt) {
        const float* mp = &mask[(size_t)q * SS + kb + kt * 16 + hf * 8];
        const float4 m0v = *(const float4*)(mp);
        const float4 m1v = *(const float4*)(mp + 4);
        const float mv[8] = {m0v.x, m0v.y, m0v.z, m0v.w, m1v.x, m1v.y, m1v.z, m1v.w};
#pragma unroll
        for (int j = 0; j < 8; ++j) {
          p[kt][j] = st[kt][qt][j] * 0.125f + mv[j];
          cm = fmaxf(cm, p[kt][j]);
        }
      }
      cm = fmaxf(cm, __shfl_xor(cm, 16, 32));   // combine the two key-halves
      const float mn    = fmaxf(mrun[qt], cm);
      const float alpha = __expf(mrun[qt] - mn);
      mrun[qt] = mn;
      float ls = 0.f;
#pragma unroll
      for (int kt = 0; kt < 2; ++kt)
#pragma unroll
        for (int j = 0; j < 8; ++j) {
          p[kt][j] = __expf(p[kt][j] - mn);
          ls += p[kt][j];
        }
      ls += __shfl_xor(ls, 16, 32);
      lrun[qt] = lrun[qt] * alpha + ls;
      // rescale O^T (queries in lanes -> plain per-lane scalar multiply)
#pragma unroll
      for (int dt = 0; dt < 4; ++dt)
#pragma unroll
        for (int j = 0; j < 8; ++j) ot[qt][dt][j] *= alpha;

      // P^T tile -> LDS [query][key], keys contiguous: one b128 per kt
      unsigned short* L = plds[wib][qt] + r * 32;
#pragma unroll
      for (int kt = 0; kt < 2; ++kt) {
        PackBF pk;
#pragma unroll
        for (int j = 0; j < 8; ++j) pk.us[j] = f2bf(p[kt][j]);
        *(uint4*)&L[kt * 16 + hf * 8] = pk.u;
      }
    }
    asm volatile("s_wait_dscnt 0" ::: "memory");  // intra-wave LDS RAW

    // P^T B-fragments: lane=query col, key chunks hf*8 and 16+hf*8
    FragBF pb[2];
#pragma unroll
    for (int qt = 0; qt < 2; ++qt) {
      const unsigned short* lp = plds[wib][qt] + r * 32 + hf * 8;
      pb[qt].u[0] = *(const uint4*)(lp);
      pb[qt].u[1] = *(const uint4*)(lp + 16);
    }

    // ---- O^T += V^T(A) x P^T(B) : V^T fragments shared by both q tiles ----
#pragma unroll
    for (int dt = 0; dt < 4; ++dt) {
      FragBF va;
      // A 16x32: lane = d row, key chunks kb+hf*8 and kb+16+hf*8
      const unsigned short* vp = Vp + (size_t)(dt * 16 + r) * SS + kb + hf * 8;
      va.u[0] = *(const uint4*)(vp);
      va.u[1] = *(const uint4*)(vp + 16);
#pragma unroll
      for (int qt = 0; qt < 2; ++qt)
        ot[qt][dt] = __builtin_amdgcn_wmma_f32_16x16x32_bf16(
            false, va.v, false, pb[qt].v, (short)0, ot[qt][dt], false, false);
    }
  }

  // ---- normalize + write AO [m=b*S+q, h*64+d]; d contiguous -> b128 stores
  const int b = bh / NH, h = bh % NH;
#pragma unroll
  for (int qt = 0; qt < 2; ++qt) {
    const float inv = 1.0f / lrun[qt];
    const int m = b * SS + q0 + qt * 16 + r;
#pragma unroll
    for (int dt = 0; dt < 4; ++dt) {
      PackBF pk;
#pragma unroll
      for (int j = 0; j < 8; ++j) pk.us[j] = f2bf(ot[qt][dt][j] * inv);
      *(uint4*)&AO[(size_t)m * HID + h * HD + dt * 16 + hf * 8] = pk.u;
    }
  }
}

// ------------------------------- launch ------------------------------------
extern "C" void kernel_launch(void* const* d_in, const int* in_sizes, int n_in,
                              void* d_out, int out_size, void* d_ws, size_t ws_size,
                              hipStream_t stream) {
  const float* x    = (const float*)d_in[0];
  const float* mask = (const float*)d_in[1];
  const float* Wq   = (const float*)d_in[2];
  const float* Wk   = (const float*)d_in[3];
  const float* Wv   = (const float*)d_in[4];
  const float* Wo   = (const float*)d_in[5];
  float* out = (float*)d_out;

  char* ws = (char*)d_ws;
  unsigned short* xb  = (unsigned short*)(ws + 0);                    // 8 MB
  unsigned short* wqb = (unsigned short*)(ws + (8u  << 20));          // 2 MB
  unsigned short* wkb = (unsigned short*)(ws + (10u << 20));          // 2 MB
  unsigned short* wvb = (unsigned short*)(ws + (12u << 20));          // 2 MB
  unsigned short* wob = (unsigned short*)(ws + (14u << 20));          // 2 MB
  unsigned short* Qb  = (unsigned short*)(ws + (16u << 20));          // 8 MB
  unsigned short* Kb  = (unsigned short*)(ws + (24u << 20));          // 8 MB
  unsigned short* Vt  = (unsigned short*)(ws + (32u << 20));          // 8 MB
  unsigned short* AO  = (unsigned short*)(ws + (40u << 20));          // 8 MB

  const int nx4 = MTOT * HID / 4;      // 1,048,576
  const int nw4 = HID * HID / 4;       //   262,144
  cvt_f32_bf16<<<(nx4 + 255) / 256, 256, 0, stream>>>(x,  xb,  nx4);
  cvt_f32_bf16<<<(nw4 + 255) / 256, 256, 0, stream>>>(Wq, wqb, nw4);
  cvt_f32_bf16<<<(nw4 + 255) / 256, 256, 0, stream>>>(Wk, wkb, nw4);
  cvt_f32_bf16<<<(nw4 + 255) / 256, 256, 0, stream>>>(Wv, wvb, nw4);
  cvt_f32_bf16<<<(nw4 + 255) / 256, 256, 0, stream>>>(Wo, wob, nw4);

  // 64x64 tile per wave: (4096/64)*(1024/64) = 1024 waves = 128 blocks x 8
  const int gemmBlocks = (MTOT / 64) * (HID / 64) / 8;
  gemm_bf16<1><<<gemmBlocks, 256, 0, stream>>>(xb, wqb, Qb,  MTOT, HID, HID);
  gemm_bf16<1><<<gemmBlocks, 256, 0, stream>>>(xb, wkb, Kb,  MTOT, HID, HID);
  gemm_bf16<2><<<gemmBlocks, 256, 0, stream>>>(xb, wvb, Vt,  MTOT, HID, HID);

  // 32 queries per wave: 2*16*(2048/32) = 2048 waves = 256 blocks x 8
  attn_kernel<<<(BB * NH * (SS / 32)) / 8, 256, 0, stream>>>(Qb, Kb, Vt, mask, AO);

  gemm_bf16<0><<<gemmBlocks, 256, 0, stream>>>(AO, wob, out, MTOT, HID, HID);
}
